// SwinTransformer_40321152975592
// MI455X (gfx1250) — matrix operand order repaired and checked by reading, not compiled, not run on hardware
//
#include <hip/hip_runtime.h>

typedef float v8f __attribute__((ext_vector_type(8)));
typedef __bf16 v16bf __attribute__((ext_vector_type(16)));

union Frag {                    // one WMMA 16-bit operand: 16 elems = 32 bytes
    v16bf v;
    uint4 q[2];
};

__device__ __forceinline__ unsigned short f2bf(float f) {
    unsigned int u = __float_as_uint(f);
    u += 0x7FFFu + ((u >> 16) & 1u);          // round-to-nearest-even
    return (unsigned short)(u >> 16);
}

// ---------------------------------------------------------------------------
// Weight f32 -> bf16 conversion
// ---------------------------------------------------------------------------
__global__ void cvt_bf16(const float* __restrict__ src,
                         unsigned short* __restrict__ dst, int n) {
    int i = blockIdx.x * 256 + threadIdx.x;
    if (i < n) dst[i] = f2bf(src[i]);
}

// ---------------------------------------------------------------------------
// Patch embed: 4x4 stride-4 conv, 3->128 ch. One block per token, thread = ch.
// ---------------------------------------------------------------------------
__global__ __launch_bounds__(128) void patch_embed(
    const float* __restrict__ x, const float* __restrict__ cw,
    const float* __restrict__ cb, unsigned short* __restrict__ tout) {
    __shared__ float sx[48];
    int token = blockIdx.x;
    int b  = token / 12544;
    int hw = token - b * 12544;
    int ph = hw / 112, pw = hw - ph * 112;
    int tid = threadIdx.x;
    if (tid < 48) {
        int ci = tid >> 4, k = tid & 15, kh = k >> 2, kw = k & 3;
        sx[tid] = x[((size_t)(b * 3 + ci) * 448 + ph * 4 + kh) * 448 + pw * 4 + kw];
    }
    __syncthreads();
    float s = cb[tid];
#pragma unroll
    for (int k2 = 0; k2 < 48; ++k2) s += sx[k2] * cw[tid * 48 + k2];
    tout[(size_t)token * 128 + tid] = f2bf(s);
}

// ---------------------------------------------------------------------------
// Token GEMM: out[M][Ntot] = A[M][128] @ W^T + bias.  A bf16 [M][K],
// W bf16 (out,in) -> staged untransposed as [N][K] (matches this code's B-fragment
// layout: per lane K=16h..16h+15 contiguous).  Block tile 128x64, 8 waves
// (4x2), wave tile 32x32.  mode 0: bf16 out [token][Ntot]; mode 1: fp32 NCHW.
// ---------------------------------------------------------------------------
__global__ __launch_bounds__(256) void gemm_bf16(
    const unsigned short* __restrict__ A, const unsigned short* __restrict__ W,
    const float* __restrict__ bias, unsigned short* __restrict__ outBf,
    float* __restrict__ outF, int Ntot, int mode) {
    const int K = 128;
    __shared__ unsigned short sA[128 * 40];   // [128][32] pad->40 (16B-aligned rows)
    __shared__ unsigned short sB[64 * 40];    // [64][32]  [n][k] layout
    int m0 = blockIdx.x * 128;
    int n0 = blockIdx.y * 64;
    int tid = threadIdx.x;
    int lane = tid & 31, wave = tid >> 5;
    int wm = wave >> 1, wn = wave & 1;
    int hh = lane >> 4, ln = lane & 15;

    v8f acc[2][2];
#pragma unroll
    for (int i = 0; i < 2; ++i)
#pragma unroll
        for (int j = 0; j < 2; ++j) acc[i][j] = {};

    for (int k0 = 0; k0 < K; k0 += 32) {
        // stage A: 128 rows x 4 uint4 (b128 in / b128 out)
#pragma unroll
        for (int idx = tid; idx < 512; idx += 256) {
            int row = idx >> 2, q = idx & 3;
            *(uint4*)(sA + row * 40 + q * 8) =
                *(const uint4*)(A + (size_t)(m0 + row) * K + k0 + q * 8);
        }
        // stage B: 64 rows x 4 uint4, straight copy (no transpose)
        {
            int nn = tid >> 2, q = tid & 3;
            *(uint4*)(sB + nn * 40 + q * 8) =
                *(const uint4*)(W + (size_t)(n0 + nn) * K + k0 + q * 8);
        }
        __syncthreads();

        Frag bf[2];
#pragma unroll
        for (int nt = 0; nt < 2; ++nt) {
            const unsigned short* p = sB + (wn * 32 + nt * 16 + ln) * 40 + 16 * hh;
            bf[nt].q[0] = *(const uint4*)p;
            bf[nt].q[1] = *(const uint4*)(p + 8);
        }
#pragma unroll
        for (int mt = 0; mt < 2; ++mt) {
            Frag af;
            const unsigned short* p = sA + (wm * 32 + mt * 16 + ln) * 40 + 16 * hh;
            af.q[0] = *(const uint4*)p;
            af.q[1] = *(const uint4*)(p + 8);
#pragma unroll
            for (int nt = 0; nt < 2; ++nt)
                acc[mt][nt] = __builtin_amdgcn_wmma_f32_16x16x32_bf16(
                    false, af.v, false, bf[nt].v, (short)0, acc[mt][nt], false, false);
        }
        __syncthreads();
    }
    // epilogue (whole 128-row block lies in one image: 12544 = 98*128)
    int bimg = m0 / 12544;
    int hw0  = m0 - bimg * 12544;
#pragma unroll
    for (int mt = 0; mt < 2; ++mt)
#pragma unroll
        for (int nt = 0; nt < 2; ++nt) {
            int col = n0 + wn * 32 + nt * 16 + ln;
            float bv = bias[col];
#pragma unroll
            for (int r = 0; r < 8; ++r) {
                int row = wm * 32 + mt * 16 + r + 8 * hh;
                float v = acc[mt][nt][r] + bv;
                if (mode == 0) {
                    outBf[(size_t)(m0 + row) * Ntot + col] = f2bf(v);
                } else {
                    outF[((size_t)(bimg * 128 + col)) * 12544 + hw0 + row] = v;
                }
            }
        }
}

// ---------------------------------------------------------------------------
// Windowed attention: one block per window (b, wh, ww), one wave per head.
// qkv: [T][384] bf16 (q|k|v, each [NH=4][hd=32]).  outa: [T][128] bf16.
// ---------------------------------------------------------------------------
__device__ __forceinline__ int regf(int p) {      // H=112, ws=7, shift=3
    return p < 105 ? 0 : (p < 109 ? 1 : 2);
}

__global__ __launch_bounds__(128) void swin_attn(
    const unsigned short* __restrict__ qkv, const float* __restrict__ rpb,
    unsigned short* __restrict__ outa, int shift) {
    __shared__ unsigned short sQK[16384];   // sQ[64][128] | sK[64][128]; reused as sP 4x[64][64]
    __shared__ unsigned short sVt[128 * 72]; // V transposed: [dim][token] pad->72
    __shared__ float sBias[676];             // rpb staged: [169][4]

    unsigned short* sQ = sQK;
    unsigned short* sK = sQK + 8192;

    int w  = blockIdx.x;
    int b  = w >> 8;
    int wi = w & 255;
    int wh = wi >> 4, ww = wi & 15;
    int tid = threadIdx.x;
    int lane = tid & 31, wave = tid >> 5;
    int hh = lane >> 4, ln = lane & 15;
    int head = wave, hdoff = head * 32;

    for (int i = tid; i < 676; i += 128) sBias[i] = rpb[i];

    // gather q/k/v for the 49 window tokens (rolled by -shift), zero-pad to 64
    // 64 rows x 48 uint4; V is stored transposed for the AV B-operand.
    for (int idx = tid; idx < 64 * 48; idx += 128) {
        int row = idx / 48;
        int d4  = idx - row * 48;
        int sec = d4 >> 4, c4 = d4 & 15;
        uint4 val = make_uint4(0u, 0u, 0u, 0u);
        if (row < 49) {
            int r = row / 7, c = row - r * 7;
            int hg = wh * 7 + r + shift; if (hg >= 112) hg -= 112;
            int wg = ww * 7 + c + shift; if (wg >= 112) wg -= 112;
            int token = b * 12544 + hg * 112 + wg;
            val = *(const uint4*)(qkv + (size_t)token * 384 + sec * 128 + c4 * 8);
        }
        if (sec < 2) {
            unsigned short* dst = (sec == 0) ? sQ : sK;
            *(uint4*)(dst + row * 128 + c4 * 8) = val;
        } else {
            unsigned int uu[4] = {val.x, val.y, val.z, val.w};
#pragma unroll
            for (int t = 0; t < 4; ++t) {
                int dim = c4 * 8 + t * 2;
                sVt[dim * 72 + row]       = (unsigned short)(uu[t] & 0xFFFFu);
                sVt[(dim + 1) * 72 + row] = (unsigned short)(uu[t] >> 16);
            }
        }
    }
    __syncthreads();

    // ---- S = Q Kt : 4x4 tiles, K=32 in one WMMA each ----
    Frag bk[4];
#pragma unroll
    for (int nt = 0; nt < 4; ++nt) {
        const unsigned short* p = sK + (nt * 16 + ln) * 128 + hdoff + 16 * hh;
        bk[nt].q[0] = *(const uint4*)p;
        bk[nt].q[1] = *(const uint4*)(p + 8);
    }
    v8f S[4][4];
#pragma unroll
    for (int mt = 0; mt < 4; ++mt) {
        Frag aq;
        const unsigned short* p = sQ + (mt * 16 + ln) * 128 + hdoff + 16 * hh;
        aq.q[0] = *(const uint4*)p;
        aq.q[1] = *(const uint4*)(p + 8);
#pragma unroll
        for (int nt = 0; nt < 4; ++nt) {
            S[mt][nt] = {};
            S[mt][nt] = __builtin_amdgcn_wmma_f32_16x16x32_bf16(
                false, aq.v, false, bk[nt].v, (short)0, S[mt][nt], false, false);
        }
    }
    __syncthreads();   // all waves done reading sQ/sK -> safe to alias as sP

    // ---- bias + mask + softmax, write probs bf16 into sP ----
    const float scale = 0.17677669529663687f;   // 1/sqrt(32)
    unsigned short* sPh = sQK + head * 4096;    // [64][64]
    bool shifted = shift > 0;
#pragma unroll
    for (int mt = 0; mt < 4; ++mt) {
#pragma unroll
        for (int r = 0; r < 8; ++r) {
            int i  = mt * 16 + r + 8 * hh;
            int ri = i / 7, ci = i - ri * 7;
            int regi = 0;
            if (shifted && i < 49)
                regi = regf(wh * 7 + ri) * 3 + regf(ww * 7 + ci);
            float vals[4];
#pragma unroll
            for (int nt = 0; nt < 4; ++nt) {
                int j = nt * 16 + ln;
                float s = S[mt][nt][r] * scale;
                if (j < 49 && i < 49) {
                    int rj = j / 7, cj = j - rj * 7;
                    s += sBias[((ri - rj + 6) * 13 + (ci - cj + 6)) * 4 + head];
                    if (shifted) {
                        int regj = regf(wh * 7 + rj) * 3 + regf(ww * 7 + cj);
                        if (regj != regi) s -= 100.0f;
                    }
                }
                if (j >= 49) s = -1e30f;
                vals[nt] = s;
            }
            float mx = fmaxf(fmaxf(vals[0], vals[1]), fmaxf(vals[2], vals[3]));
#pragma unroll
            for (int off = 8; off >= 1; off >>= 1)
                mx = fmaxf(mx, __shfl_xor(mx, off, 32));
            float sum = 0.f;
#pragma unroll
            for (int nt = 0; nt < 4; ++nt) {
                vals[nt] = __expf(vals[nt] - mx);
                sum += vals[nt];
            }
#pragma unroll
            for (int off = 8; off >= 1; off >>= 1)
                sum += __shfl_xor(sum, off, 32);
            float inv = 1.0f / sum;
#pragma unroll
            for (int nt = 0; nt < 4; ++nt)
                sPh[i * 64 + nt * 16 + ln] = f2bf(vals[nt] * inv);
        }
    }

    // ---- O = P @ V : 4(M) x 2(N=32) tiles, K=64 in two steps; scatter back ----
#pragma unroll
    for (int mt = 0; mt < 4; ++mt) {
#pragma unroll
        for (int nt2 = 0; nt2 < 2; ++nt2) {
            v8f o = {};
#pragma unroll
            for (int ks = 0; ks < 2; ++ks) {
                Frag ap, bv;
                const unsigned short* pa = sPh + (mt * 16 + ln) * 64 + ks * 32 + 16 * hh;
                ap.q[0] = *(const uint4*)pa;
                ap.q[1] = *(const uint4*)(pa + 8);
                const unsigned short* pb =
                    sVt + (hdoff + nt2 * 16 + ln) * 72 + ks * 32 + 16 * hh;
                bv.q[0] = *(const uint4*)pb;
                bv.q[1] = *(const uint4*)(pb + 8);
                o = __builtin_amdgcn_wmma_f32_16x16x32_bf16(
                    false, ap.v, false, bv.v, (short)0, o, false, false);
            }
#pragma unroll
            for (int r = 0; r < 8; ++r) {
                int i = mt * 16 + r + 8 * hh;
                if (i < 49) {
                    int ri = i / 7, ci = i - ri * 7;
                    int hg = wh * 7 + ri + shift; if (hg >= 112) hg -= 112;
                    int wg = ww * 7 + ci + shift; if (wg >= 112) wg -= 112;
                    size_t token = (size_t)b * 12544 + hg * 112 + wg;
                    outa[token * 128 + hdoff + nt2 * 16 + ln] = f2bf(o[r]);
                }
            }
        }
    }
}

// ---------------------------------------------------------------------------
extern "C" void kernel_launch(void* const* d_in, const int* in_sizes, int n_in,
                              void* d_out, int out_size, void* d_ws, size_t ws_size,
                              hipStream_t stream) {
    const float* x      = (const float*)d_in[0];
    const float* conv_w = (const float*)d_in[1];
    const float* conv_b = (const float*)d_in[2];
    const float* qkv_w  = (const float*)d_in[3];
    const float* qkv_b  = (const float*)d_in[4];
    const float* proj_w = (const float*)d_in[5];
    const float* proj_b = (const float*)d_in[6];
    const float* rpb    = (const float*)d_in[7];
    float* out = (float*)d_out;

    const int T = 200704;   // 16 * 112 * 112 tokens
    char* ws = (char*)d_ws;
    size_t off = 0;
    auto carve = [&](size_t bytes) {
        char* p = ws + off;
        off = (off + bytes + 255) & ~(size_t)255;
        return p;
    };
    unsigned short* tbuf  = (unsigned short*)carve((size_t)T * 128 * 2);
    unsigned short* qkvb  = (unsigned short*)carve((size_t)T * 384 * 2);
    unsigned short* abuf  = (unsigned short*)carve((size_t)T * 128 * 2);
    unsigned short* wqkv  = (unsigned short*)carve((size_t)2 * 384 * 128 * 2);
    unsigned short* wproj = (unsigned short*)carve((size_t)2 * 128 * 128 * 2);

    cvt_bf16<<<(2 * 384 * 128 + 255) / 256, 256, 0, stream>>>(qkv_w, wqkv, 2 * 384 * 128);
    cvt_bf16<<<(2 * 128 * 128 + 255) / 256, 256, 0, stream>>>(proj_w, wproj, 2 * 128 * 128);
    patch_embed<<<T, 128, 0, stream>>>(x, conv_w, conv_b, tbuf);

    for (int blk = 0; blk < 2; ++blk) {
        int shift = blk ? 3 : 0;
        gemm_bf16<<<dim3(1568, 6), 256, 0, stream>>>(
            tbuf, wqkv + (size_t)blk * 384 * 128, qkv_b + blk * 384, qkvb,
            nullptr, 384, 0);
        swin_attn<<<4096, 128, 0, stream>>>(qkvb, rpb + blk * 169 * 4, abuf, shift);
        if (blk == 0)
            gemm_bf16<<<dim3(1568, 2), 256, 0, stream>>>(
                abuf, wproj, proj_b, tbuf, nullptr, 128, 0);
        else
            gemm_bf16<<<dim3(1568, 2), 256, 0, stream>>>(
                abuf, wproj + 128 * 128, proj_b + 128, nullptr, out, 128, 1);
    }
}
